// TemporalContextBlock_84318797955512
// MI455X (gfx1250) — matrix operand order repaired and checked by reading, not compiled, not run on hardware
//
#include <hip/hip_runtime.h>
#include <hip/hip_bf16.h>

typedef __attribute__((ext_vector_type(16))) _Float16 v16h;
typedef __attribute__((ext_vector_type(8)))  _Float16 v8h;
typedef __attribute__((ext_vector_type(8)))  float    v8f;

#define BATCH 4
#define SLOTS 8
#define CH    256
#define PIX   4096   // 64*64
#define EPSV  1e-5f

// fm slot ordering: s'=0..6 -> x slots {0,1,2,4,5,6,7}, s'=7 -> x slot 3 (main)
__device__ __forceinline__ int srcSlot(int s) { return s < 3 ? s : (s < 7 ? s + 1 : 3); }

__device__ __forceinline__ float blockReduce256(float v, float* sm) {
    int tid = threadIdx.x;
    sm[tid] = v; __syncthreads();
    for (int off = 128; off > 0; off >>= 1) {
        if (tid < off) sm[tid] += sm[tid + off];
        __syncthreads();
    }
    float r = sm[0]; __syncthreads();
    return r;
}

// ---------------------------------------------------------------- init stats
__global__ void k_init(float* sums, float* bcs1, float* bcs2, float* gstats) {
    int t = threadIdx.x;
    for (int i = t; i < BATCH * CH; i += 256) { bcs1[i] = 0.f; bcs2[i] = 0.f; }
    if (t < BATCH * SLOTS) sums[t] = 0.f;
    if (t < 2 * BATCH)     gstats[t] = 0.f;
}

// ------------------------------------------------- f32 -> f16 weight prep
__global__ void k_prep(const float* __restrict__ W3o, const float* __restrict__ b3o,
                       const float* __restrict__ W3c, const float* __restrict__ b3c,
                       const float* __restrict__ Wg,
                       _Float16* __restrict__ Wcomb, _Float16* __restrict__ Wg16,
                       float* __restrict__ bias3) {
    int o = blockIdx.x, t = threadIdx.x;
    Wcomb[(size_t)o * 512 + t]       = (_Float16)W3o[(size_t)o * CH + t];
    Wcomb[(size_t)o * 512 + 256 + t] = (_Float16)W3c[(size_t)o * CH + t];
    Wg16[(size_t)o * CH + t]         = (_Float16)Wg[(size_t)o * CH + t];
    if (t == 0) bias3[o] = 7.f * b3o[o] + b3c[o];
}

// ------------------------------------------------------- step1 logits (1xC)
__global__ void k1_logits(const float* __restrict__ x, const float* __restrict__ W1o,
                          const float* __restrict__ b1o, const float* __restrict__ W1c,
                          const float* __restrict__ b1c, float* __restrict__ l) {
    int p  = blockIdx.x * 256 + threadIdx.x;
    int bs = blockIdx.y; int b = bs >> 3, s = bs & 7;
    const float* Wv = (s < 7) ? W1o : W1c;
    float acc = (s < 7) ? b1o[0] : b1c[0];
    const float* xp = x + ((size_t)(b * SLOTS + srcSlot(s)) * CH) * PIX + p;
    for (int c = 0; c < CH; ++c) acc += Wv[c] * xp[(size_t)c * PIX];
    l[(size_t)bs * PIX + p] = acc;
}

// --------------------------------- softmax over S per (b,p) + per-(b,s) sums
__global__ void k2a_softmax(const float* __restrict__ l, float* __restrict__ sm,
                            float* __restrict__ sums) {
    __shared__ float ssum[SLOTS];
    int p = blockIdx.x * 256 + threadIdx.x;
    int b = blockIdx.y;
    if (threadIdx.x < SLOTS) ssum[threadIdx.x] = 0.f;
    __syncthreads();
    float v[SLOTS]; float mx = -1e30f;
    for (int s = 0; s < SLOTS; ++s) {
        v[s] = l[(size_t)(b * SLOTS + s) * PIX + p];
        mx = fmaxf(mx, v[s]);
    }
    float tot = 0.f;
    for (int s = 0; s < SLOTS; ++s) { v[s] = __expf(v[s] - mx); tot += v[s]; }
    float inv = 1.f / tot;
    for (int s = 0; s < SLOTS; ++s) {
        float sv = v[s] * inv;
        sm[(size_t)(b * SLOTS + s) * PIX + p] = sv;
        atomicAdd(&ssum[s], sv);
    }
    __syncthreads();
    if (threadIdx.x < SLOTS) atomicAdd(&sums[b * SLOTS + threadIdx.x], ssum[threadIdx.x]);
}

// ------------------------------------------- a = softmax * mean_p(softmax)
__global__ void k2b_scale(const float* __restrict__ sm, const float* __restrict__ sums,
                          float* __restrict__ a) {
    int p  = blockIdx.x * 256 + threadIdx.x;
    int bs = blockIdx.y;
    float mean = sums[bs] * (1.f / PIX);
    a[(size_t)bs * PIX + p] = sm[(size_t)bs * PIX + p] * mean;
}

// ---------------------------- y[b,s,c] = sum_p x[b,src(s),c,p] * a[b,s,p]
__global__ void k3a_y(const float* __restrict__ x, const float* __restrict__ a,
                      float* __restrict__ y) {
    __shared__ float red[256];
    int bsc = blockIdx.x;
    int c = bsc & 255; int bs = bsc >> 8; int b = bs >> 3, s = bs & 7;
    const float* xp = x + (size_t)((b * SLOTS + srcSlot(s)) * CH + c) * PIX;
    const float* ap = a + (size_t)bs * PIX;
    float acc = 0.f;
    for (int p = threadIdx.x; p < PIX; p += 256) acc += xp[p] * ap[p];
    acc = blockReduce256(acc, red);
    if (threadIdx.x == 0) y[bsc] = acc;
}

// ---------------- agg[b,s,c] = W2sel[c,:] . y[b,s,:] + b2sel[c]*sum_p(a)
__global__ void k3b_agg(const float* __restrict__ y, const float* __restrict__ sums,
                        const float* __restrict__ W2o, const float* __restrict__ b2o,
                        const float* __restrict__ W2c, const float* __restrict__ b2c,
                        float* __restrict__ agg) {
    __shared__ float sy[CH];
    int bs = blockIdx.x; int s = bs & 7;
    int c = threadIdx.x;
    sy[c] = y[(size_t)bs * CH + c];
    __syncthreads();
    const float* W  = (s < 7) ? W2o : W2c;
    const float* bb = (s < 7) ? b2o : b2c;
    const float* Wr = W + (size_t)c * CH;
    float acc = 0.f;
    for (int cc = 0; cc < CH; ++cc) acc += Wr[cc] * sy[cc];
    float sv = sums[bs];                     // sum_p a = (sum_p sm)^2 / P
    agg[(size_t)bs * CH + c] = acc + bb[c] * (sv * sv * (1.f / PIX));
}

// ----- fm = x+agg; accumulate per-(b,c) moments; u=sum_{s<7} fm; v=fm[s=7]
__global__ void k4a_fm(const float* __restrict__ x, const float* __restrict__ agg,
                       float* __restrict__ u_raw, float* __restrict__ v_raw,
                       float* __restrict__ bcs1, float* __restrict__ bcs2) {
    __shared__ float red[256];
    int p  = blockIdx.x * 256 + threadIdx.x;
    int bc = blockIdx.y; int b = bc >> 8, c = bc & 255;
    float u = 0.f, vv = 0.f, s1 = 0.f, s2 = 0.f;
    for (int s = 0; s < SLOTS; ++s) {
        float ag = agg[(size_t)(b * SLOTS + s) * CH + c];
        float f  = x[(size_t)((b * SLOTS + srcSlot(s)) * CH + c) * PIX + p] + ag;
        s1 += f; s2 += f * f;
        if (s < 7) u += f; else vv = f;
    }
    u_raw[(size_t)bc * PIX + p] = u;
    v_raw[(size_t)bc * PIX + p] = vv;
    float r1 = blockReduce256(s1, red);
    float r2 = blockReduce256(s2, red);
    if (threadIdx.x == 0) { atomicAdd(&bcs1[bc], r1); atomicAdd(&bcs2[bc], r2); }
}

// ------------------------------------------------ per-(b,c) mu / rstd
__global__ void k4b_mustats(const float* __restrict__ bcs1, const float* __restrict__ bcs2,
                            float* __restrict__ mu, float* __restrict__ rstd) {
    int i = blockIdx.x * 256 + threadIdx.x;        // 0..1023
    float m   = bcs1[i] * (1.f / (SLOTS * PIX));
    float var = bcs2[i] * (1.f / (SLOTS * PIX)) - m * m;
    mu[i] = m; rstd[i] = rsqrtf(var + EPSV);
}

// --------- build normalized f16 activations, K-contiguous: act[b][p][512]
__global__ void k4c_act(const float* __restrict__ u_raw, const float* __restrict__ v_raw,
                        const float* __restrict__ mu, const float* __restrict__ rstd,
                        _Float16* __restrict__ act) {
    int bp = blockIdx.x; int b = bp >> 12, p = bp & (PIX - 1);
    int c  = threadIdx.x;
    int bc = b * CH + c;
    float m = mu[bc], r = rstd[bc];
    float un = (u_raw[(size_t)bc * PIX + p] - 7.f * m) * r;
    float vn = (v_raw[(size_t)bc * PIX + p] - m) * r;
    _Float16* ap = act + (size_t)bp * 512;
    ap[c]        = (_Float16)un;
    ap[256 + c]  = (_Float16)vn;
}

// ------------------------------------------------------------- WMMA GEMM
// D[b][m][n] = sum_k A[m][k]*Bact[b][n][k] + bias[m]   (m,n in 256 x 4096)
// Output stored transposed [b][n][m] so the next GEMM's B is K-contiguous.
template<int KDIM, bool RELU, bool HALF_OUT, bool STATS>
__global__ __launch_bounds__(256) void k_gemm(
        const _Float16* __restrict__ A,     // [256][KDIM] f16
        const _Float16* __restrict__ Bact,  // [BATCH][PIX][KDIM] f16
        const float*    __restrict__ bias,  // [256]
        _Float16*       __restrict__ outH,  // [BATCH][PIX][256] f16  (HALF_OUT)
        float*          __restrict__ outF,  // [BATCH][PIX][256] f32  (!HALF_OUT)
        float*          __restrict__ gstats) {
    __shared__ __align__(16) _Float16 sA[128][40];
    __shared__ __align__(16) _Float16 sB[128][40];
    __shared__ float sBias[128];

    const int b   = blockIdx.z;
    const int m0  = blockIdx.y * 128;
    const int n0  = blockIdx.x * 128;
    const int tid = threadIdx.x;
    const int lane = tid & 31;
    const int wave = tid >> 5;
    const int wm = wave >> 2;               // 0..1 : 64-row half
    const int wn = wave & 3;                // 0..3 : 32-col quarter

    if (tid < 128) sBias[tid] = bias[m0 + tid];

    const _Float16* Bb = Bact + ((size_t)b * PIX + n0) * KDIM;

    const v8f vzero = {0.f, 0.f, 0.f, 0.f, 0.f, 0.f, 0.f, 0.f};
    v8f acc[4][2];
    for (int mt = 0; mt < 4; ++mt)
        for (int nt = 0; nt < 2; ++nt) acc[mt][nt] = vzero;

    const int klo = (lane >> 4) * 8;        // A frag: lanes 0-15 k0..7/16..23, 16-31 k8..15/24..31
    const int kkB = (lane >> 4) * 8;        // B frag: lanes 0-15 k0..7/16..23, 16-31 k8..15/24..31

    for (int k0 = 0; k0 < KDIM; k0 += 32) {
        // stage 128x32 f16 tiles of A (weights) and B (activations, n-major)
        for (int i = 0; i < 2; ++i) {
            int q  = tid + i * 256;          // 0..511
            int r  = q >> 2;                 // row 0..127
            int cc = (q & 3) * 8;            // k-sub 0,8,16,24
            *(int4*)&sA[r][cc] = *(const int4*)&A [(size_t)(m0 + r) * KDIM + k0 + cc];
            *(int4*)&sB[r][cc] = *(const int4*)&Bb[(size_t)r        * KDIM + k0 + cc];
        }
        __syncthreads();

        union Frag { int4 q[2]; v16h v; };
        Frag bf[2];
        for (int nt = 0; nt < 2; ++nt) {
            int n = wn * 32 + nt * 16 + (lane & 15);
            bf[nt].q[0] = *(const int4*)&sB[n][kkB];
            bf[nt].q[1] = *(const int4*)&sB[n][kkB + 16];
        }
        for (int mt = 0; mt < 4; ++mt) {
            int m = wm * 64 + mt * 16 + (lane & 15);
            Frag af;
            af.q[0] = *(const int4*)&sA[m][klo];
            af.q[1] = *(const int4*)&sA[m][klo + 16];
            for (int nt = 0; nt < 2; ++nt) {
                acc[mt][nt] = __builtin_amdgcn_wmma_f32_16x16x32_f16(
                    false, af.v, false, bf[nt].v, (short)0, acc[mt][nt], false, false);
            }
        }
        __syncthreads();
    }

    // epilogue: bias (+relu), transposed b128 stores, optional stats
    float lsum = 0.f, lsq = 0.f;
    for (int mt = 0; mt < 4; ++mt) {
        int mBase = wm * 64 + mt * 16 + (lane >> 4) * 8;   // D: vgpr i -> m = mBase+i
        for (int nt = 0; nt < 2; ++nt) {
            int n = n0 + wn * 32 + nt * 16 + (lane & 15);
            float vals[8];
            for (int i = 0; i < 8; ++i) {
                float v = acc[mt][nt][i] + sBias[mBase + i];
                if (RELU) v = fmaxf(v, 0.f);
                if (STATS) { lsum += v; lsq += v * v; }
                vals[i] = v;
            }
            size_t base = ((size_t)b * PIX + n) * 256 + (m0 + mBase);
            if (HALF_OUT) {
                v8h hv;
                for (int i = 0; i < 8; ++i) hv[i] = (_Float16)vals[i];
                *(v8h*)&outH[base] = hv;
            } else {
                *(float4*)&outF[base]     = make_float4(vals[0], vals[1], vals[2], vals[3]);
                *(float4*)&outF[base + 4] = make_float4(vals[4], vals[5], vals[6], vals[7]);
            }
        }
    }
    if (STATS) {
        __syncthreads();
        float* red = (float*)&sA[0][0];
        red[tid] = lsum; __syncthreads();
        for (int off = 128; off > 0; off >>= 1) { if (tid < off) red[tid] += red[tid + off]; __syncthreads(); }
        float bs1 = red[0]; __syncthreads();
        red[tid] = lsq; __syncthreads();
        for (int off = 128; off > 0; off >>= 1) { if (tid < off) red[tid] += red[tid + off]; __syncthreads(); }
        if (tid == 0) { atomicAdd(&gstats[b * 2], bs1); atomicAdd(&gstats[b * 2 + 1], red[0]); }
    }
}

// ----------------------- final: group-norm of g, gamma/beta, + identity
__global__ void k7_final(const float* __restrict__ gT, const float* __restrict__ gstats,
                         const float* __restrict__ gamma, const float* __restrict__ beta,
                         const float* __restrict__ x, float* __restrict__ out) {
    int p  = blockIdx.x * 256 + threadIdx.x;
    int bc = blockIdx.y; int b = bc >> 8, c = bc & 255;
    const float inv = 1.f / (float)(CH * PIX);
    float gm = gstats[b * 2] * inv;
    float gv = gstats[b * 2 + 1] * inv - gm * gm;
    float gr = rsqrtf(gv + EPSV);
    float g  = gT[((size_t)b * PIX + p) * 256 + c];
    float idv = x[(size_t)((b * SLOTS + 3) * CH + c) * PIX + p];
    out[(size_t)(b * CH + c) * PIX + p] = (g - gm) * gr * gamma[c] + beta[c] + idv;
}

extern "C" void kernel_launch(void* const* d_in, const int* in_sizes, int n_in,
                              void* d_out, int out_size, void* d_ws, size_t ws_size,
                              hipStream_t stream) {
    const float* x   = (const float*)d_in[0];
    const float* W1o = (const float*)d_in[1];
    const float* b1o = (const float*)d_in[2];
    const float* W1c = (const float*)d_in[3];
    const float* b1c = (const float*)d_in[4];
    const float* W2o = (const float*)d_in[5];
    const float* b2o = (const float*)d_in[6];
    const float* W2c = (const float*)d_in[7];
    const float* b2c = (const float*)d_in[8];
    const float* W3o = (const float*)d_in[9];
    const float* b3o = (const float*)d_in[10];
    const float* W3c = (const float*)d_in[11];
    const float* b3c = (const float*)d_in[12];
    const float* Wg  = (const float*)d_in[13];
    const float* bg  = (const float*)d_in[14];
    const float* gamma = (const float*)d_in[15];
    const float* beta  = (const float*)d_in[16];
    float* out = (float*)d_out;

    char* ws = (char*)d_ws;
    size_t off = 0;
    auto alloc = [&](size_t bytes) -> char* {
        char* p = ws + off;
        off += (bytes + 255) & ~(size_t)255;
        return p;
    };

    // region A (64 MiB): u_raw/v_raw/act early, aliased by gT (f32) later
    char* regionA = alloc(64ull << 20);
    float*    u_raw = (float*)regionA;
    float*    v_raw = (float*)(regionA + (16ull << 20));
    _Float16* act   = (_Float16*)(regionA + (32ull << 20));
    float*    gT    = (float*)regionA;     // [B][PIX][CH] f32 (after act is dead)

    float* l     = (float*)alloc((size_t)BATCH * SLOTS * PIX * 4);
    float* sm    = (float*)alloc((size_t)BATCH * SLOTS * PIX * 4);
    float* a     = (float*)alloc((size_t)BATCH * SLOTS * PIX * 4);
    float* sums  = (float*)alloc(BATCH * SLOTS * 4);
    float* y     = (float*)alloc((size_t)BATCH * SLOTS * CH * 4);
    float* agg   = (float*)alloc((size_t)BATCH * SLOTS * CH * 4);
    float* bcs1  = (float*)alloc(BATCH * CH * 4);
    float* bcs2  = (float*)alloc(BATCH * CH * 4);
    float* mu    = (float*)alloc(BATCH * CH * 4);
    float* rstd  = (float*)alloc(BATCH * CH * 4);
    _Float16* Wcomb = (_Float16*)alloc((size_t)CH * 512 * 2);
    _Float16* Wg16  = (_Float16*)alloc((size_t)CH * CH * 2);
    float*    bias3 = (float*)alloc(CH * 4);
    _Float16* out2T = (_Float16*)alloc((size_t)BATCH * PIX * CH * 2);
    float*    gstats = (float*)alloc(2 * BATCH * 4);

    const dim3 blk(256);
    k_init<<<1, blk, 0, stream>>>(sums, bcs1, bcs2, gstats);
    k_prep<<<CH, blk, 0, stream>>>(W3o, b3o, W3c, b3c, Wg, Wcomb, Wg16, bias3);
    k1_logits<<<dim3(PIX / 256, BATCH * SLOTS), blk, 0, stream>>>(x, W1o, b1o, W1c, b1c, l);
    k2a_softmax<<<dim3(PIX / 256, BATCH), blk, 0, stream>>>(l, sm, sums);
    k2b_scale<<<dim3(PIX / 256, BATCH * SLOTS), blk, 0, stream>>>(sm, sums, a);
    k3a_y<<<BATCH * SLOTS * CH, blk, 0, stream>>>(x, a, y);
    k3b_agg<<<BATCH * SLOTS, blk, 0, stream>>>(y, sums, W2o, b2o, W2c, b2c, agg);
    k4a_fm<<<dim3(PIX / 256, BATCH * CH), blk, 0, stream>>>(x, agg, u_raw, v_raw, bcs1, bcs2);
    k4b_mustats<<<BATCH, blk, 0, stream>>>(bcs1, bcs2, mu, rstd);
    k4c_act<<<BATCH * PIX, blk, 0, stream>>>(u_raw, v_raw, mu, rstd, act);
    // GEMM1: [256 x 4096] = Wcomb(256x512) @ [u;v], f16 out transposed
    k_gemm<512, false, true, false><<<dim3(PIX / 128, 2, BATCH), blk, 0, stream>>>(
        Wcomb, act, bias3, out2T, nullptr, nullptr);
    // GEMM2: g = relu(Wg(256x256) @ out + bg), f32 transposed + stats
    k_gemm<256, true, false, true><<<dim3(PIX / 128, 2, BATCH), blk, 0, stream>>>(
        Wg16, out2T, bg, nullptr, gT, gstats);
    k7_final<<<dim3(PIX / 256, BATCH * CH), blk, 0, stream>>>(gT, gstats, gamma, beta, x, out);

    (void)in_sizes; (void)n_in; (void)out_size; (void)ws_size;
}
